// LSTM_38019050504567
// MI455X (gfx1250) — matrix operand order repaired and checked by reading, not compiled, not run on hardware
//
#include <hip/hip_runtime.h>

// ---------------------------------------------------------------------------
// Seq2seq LSTM (enc 2L + dec 2L), E=512, H=1024, B=32, S=T=128, BOS=1
// Strategy:
//   * convert all weights to bf16 once (L2-resident, ~59MB < 192MB L2)
//   * per layer: P = X @ Wih^T + (bih+bhh)  -- one big WMMA GEMM [4096 x 4096]
//   * recurrence: ONE persistent kernel per layer (64 WGs, grid barrier).
//     Each step: G_partial[split] = h @ Whh[k-split]^T  (512 waves, 4-way K
//     split, v_wmma_f32_16x16x32_bf16), barrier, pointwise LSTM update
//     (sums 4 partials + P[:,t,:]), barrier, next t.
// ---------------------------------------------------------------------------

typedef __attribute__((ext_vector_type(16))) __bf16 v16bf;
typedef __attribute__((ext_vector_type(8)))  __bf16 v8bf;
typedef __attribute__((ext_vector_type(8)))  float  v8f;

#define HDIM    1024
#define GDIM    4096   // 4*H
#define SEQ     128
#define BATCH   32
#define EDIM    512
#define BOS_IDX 1

#define NWG_REC 64     // workgroups in persistent recurrent kernel
#define NSPLIT  4      // K-split factor for recurrent GEMM
#define KSPL    (HDIM / NSPLIT)   // 256

// ------------------------- fragment loaders --------------------------------
// A fragment (16x32 bf16, MxK). lane: m = lane&15, half = lane>>4.
// elems 0..7  -> K = k0 + 8*half + e ; elems 8..15 -> K = k0 + 16 + 8*half + (e-8)
__device__ __forceinline__ v16bf load_a_frag(const __bf16* __restrict__ row,
                                             int k0, int half) {
  const __bf16* p = row + k0 + 8 * half;
  v8bf lo = *(const v8bf*)(p);
  v8bf hi = *(const v8bf*)(p + 16);
  v16bf r;
#pragma unroll
  for (int i = 0; i < 8; ++i) { r[i] = lo[i]; r[i + 8] = hi[i]; }
  return r;
}

// B fragment (32x16 bf16, KxN), element (k,n) = W[n*K + k] (W is [N,K] row-major).
// lane: n = lane&15, half = lane>>4; elems 0..7 -> K = k0 + 8*half + e ; elems 8..15 -> K = k0 + 16 + 8*half + (e-8).
__device__ __forceinline__ v16bf load_b_frag(const __bf16* __restrict__ wcol,
                                             int k0, int half) {
  const __bf16* p = wcol + k0 + 8 * half;
  v8bf lo = *(const v8bf*)(p);
  v8bf hi = *(const v8bf*)(p + 16);
  v16bf r;
#pragma unroll
  for (int i = 0; i < 8; ++i) { r[i] = lo[i]; r[i + 8] = hi[i]; }
  return r;
}

__device__ __forceinline__ v8f wmma_bf16(v16bf a, v16bf b, v8f c) {
  return __builtin_amdgcn_wmma_f32_16x16x32_bf16(
      /*neg_a=*/false, a, /*neg_b=*/false, b,
      /*c_mod=*/(short)0, c, /*reuse_a=*/false, /*reuse_b=*/false);
}

__device__ __forceinline__ float sigf(float x) { return 1.0f / (1.0f + __expf(-x)); }

// ------------------------- device-scope grid barrier -----------------------
__device__ __forceinline__ void grid_barrier(unsigned* cnt, unsigned* gen,
                                             unsigned nwg) {
  __threadfence();          // make this WG's global stores visible (agent scope)
  __syncthreads();
  if (threadIdx.x == 0) {
    unsigned g = __hip_atomic_load(gen, __ATOMIC_RELAXED, __HIP_MEMORY_SCOPE_AGENT);
    unsigned arrived = __hip_atomic_fetch_add(cnt, 1u, __ATOMIC_ACQ_REL,
                                              __HIP_MEMORY_SCOPE_AGENT);
    if (arrived == nwg - 1u) {
      __hip_atomic_store(cnt, 0u, __ATOMIC_RELAXED, __HIP_MEMORY_SCOPE_AGENT);
      __hip_atomic_store(gen, g + 1u, __ATOMIC_RELEASE, __HIP_MEMORY_SCOPE_AGENT);
    } else {
      while (__hip_atomic_load(gen, __ATOMIC_ACQUIRE,
                               __HIP_MEMORY_SCOPE_AGENT) == g)
        __builtin_amdgcn_s_sleep(1);
    }
  }
  __syncthreads();
  __threadfence();          // don't let loads of barrier-protected data hoist
}

// ------------------------- input-projection GEMM ---------------------------
// Out[row, col] = sum_k A[row, k] * W[col, k] + Cadd[row*add_rs + col]
// One wave computes a 32x32 macro-tile = 2x2 WMMA tiles (4 wmma / K-step).
__global__ __launch_bounds__(256)
void k_wmma_gemm(const __bf16* __restrict__ A, int lda, int M,
                 const __bf16* __restrict__ W, int K,
                 const float* __restrict__ Cadd, long add_rs,
                 float* __restrict__ Out, long out_rs, int N) {
  const int wave = blockIdx.x * (blockDim.x >> 5) + (threadIdx.x >> 5);
  const int nt32 = N >> 5;
  const int mt = wave / nt32;
  const int nt = wave % nt32;
  const int m0 = mt * 32;
  const int n0 = nt * 32;
  if (m0 >= M) return;

  const int lane = threadIdx.x & 31;
  const int half = lane >> 4;
  const int lm   = lane & 15;

  v8f acc00 = {}, acc01 = {}, acc10 = {}, acc11 = {};

  const __bf16* arow0 = A + (size_t)(m0 + lm) * lda;
  const __bf16* arow1 = A + (size_t)(m0 + 16 + lm) * lda;
  const __bf16* wcol0 = W + (size_t)(n0 + lm) * K;
  const __bf16* wcol1 = W + (size_t)(n0 + 16 + lm) * K;

#pragma unroll 4
  for (int k0 = 0; k0 < K; k0 += 32) {
    v16bf a0 = load_a_frag(arow0, k0, half);
    v16bf a1 = load_a_frag(arow1, k0, half);
    v16bf b0 = load_b_frag(wcol0, k0, half);
    v16bf b1 = load_b_frag(wcol1, k0, half);
    acc00 = wmma_bf16(a0, b0, acc00);
    acc01 = wmma_bf16(a0, b1, acc01);
    acc10 = wmma_bf16(a1, b0, acc10);
    acc11 = wmma_bf16(a1, b1, acc11);
  }

  // D layout: VGPR r holds row (r + 8*half), col = lane&15.
  auto epi = [&](v8f& acc, int mb, int nb) {
    const int col = nb + lm;
#pragma unroll
    for (int r = 0; r < 8; ++r) {
      const int row = mb + r + 8 * half;
      float add = Cadd ? Cadd[(size_t)row * add_rs + col] : 0.0f;
      Out[(size_t)row * out_rs + col] = acc[r] + add;
    }
  };
  epi(acc00, m0,      n0);
  epi(acc01, m0,      n0 + 16);
  epi(acc10, m0 + 16, n0);
  epi(acc11, m0 + 16, n0 + 16);
}

// ------------------------- persistent recurrent layer ----------------------
// Grid: NWG_REC x 256 threads = 512 waves. Wave w -> tile = w>>2 (N columns
// 32*tile .. +31), split = w&3 (K range split*256 .. +255). Per step:
//   phase 1: Gp[split][32][4096] partial = h @ Whh^T (K-split)
//   barrier
//   phase 2: gates = sum(partials) + P[:,t,:]; LSTM pointwise; write h/c/ys
//   barrier
__global__ __launch_bounds__(256)
void k_lstm_layer(const __bf16* __restrict__ Whh,     // [4096,1024] bf16
                  const float*  __restrict__ P,       // [B*S, 4096] x@Wih^T + b
                  float* __restrict__ Gp,             // [NSPLIT][32][4096]
                  float* __restrict__ c,              // [32*1024]
                  __bf16* __restrict__ h,             // [32*1024]
                  __bf16* __restrict__ ys_bf,         // [B*S*H] or null
                  float*  __restrict__ ys_f32,        // [B*S*H] or null
                  unsigned* bar_cnt, unsigned* bar_gen) {
  const int widx  = threadIdx.x >> 5;
  const int wave  = blockIdx.x * 8 + widx;   // 0..511
  const int tile  = wave >> 2;               // 0..127  (N tile)
  const int split = wave & (NSPLIT - 1);     // 0..3    (K split)
  const int n0    = tile * 32;
  const int kbase = split * KSPL;
  const int lane  = threadIdx.x & 31;
  const int half  = lane >> 4;
  const int lm    = lane & 15;
  const int tid   = blockIdx.x * 256 + threadIdx.x;  // 0..16383

  const __bf16* arow0 = h + (size_t)lm * HDIM + kbase;
  const __bf16* arow1 = h + (size_t)(16 + lm) * HDIM + kbase;
  const __bf16* wcol0 = Whh + (size_t)(n0 + lm) * HDIM + kbase;
  const __bf16* wcol1 = Whh + (size_t)(n0 + 16 + lm) * HDIM + kbase;
  float* gout = Gp + (size_t)split * BATCH * GDIM;

  for (int t = 0; t < SEQ; ++t) {
    // ---- phase 1: partial GEMM over K range [kbase, kbase+KSPL) ----
    v8f acc00 = {}, acc01 = {}, acc10 = {}, acc11 = {};
#pragma unroll 4
    for (int kk = 0; kk < KSPL; kk += 32) {
      v16bf a0 = load_a_frag(arow0, kk, half);
      v16bf a1 = load_a_frag(arow1, kk, half);
      v16bf b0 = load_b_frag(wcol0, kk, half);
      v16bf b1 = load_b_frag(wcol1, kk, half);
      acc00 = wmma_bf16(a0, b0, acc00);
      acc01 = wmma_bf16(a0, b1, acc01);
      acc10 = wmma_bf16(a1, b0, acc10);
      acc11 = wmma_bf16(a1, b1, acc11);
    }
    {
      auto epi = [&](v8f& acc, int mb, int nb) {
        const int col = nb + lm;
#pragma unroll
        for (int r = 0; r < 8; ++r) {
          const int row = mb + r + 8 * half;
          gout[(size_t)row * GDIM + col] = acc[r];
        }
      };
      epi(acc00, 0,  n0);
      epi(acc01, 0,  n0 + 16);
      epi(acc10, 16, n0);
      epi(acc11, 16, n0 + 16);
    }
    grid_barrier(bar_cnt, bar_gen, NWG_REC);

    // ---- phase 2: pointwise LSTM (2 elements of [32 x 1024] per thread) ----
#pragma unroll
    for (int e = 0; e < 2; ++e) {
      const int i = tid + e * (NWG_REC * 256);       // 0..32767
      const int b = i >> 10;
      const int j = i & (HDIM - 1);
      const float* pr = P + ((size_t)b * SEQ + t) * GDIM;
      float gi = pr[j];
      float gf = pr[HDIM + j];
      float gg = pr[2 * HDIM + j];
      float go = pr[3 * HDIM + j];
#pragma unroll
      for (int s = 0; s < NSPLIT; ++s) {
        const float* gp = Gp + ((size_t)s * BATCH + b) * GDIM;
        gi += gp[j];
        gf += gp[HDIM + j];
        gg += gp[2 * HDIM + j];
        go += gp[3 * HDIM + j];
      }
      const float iv = sigf(gi);
      const float fv = sigf(gf);
      const float gv = tanhf(gg);
      const float ov = sigf(go);
      const float cv = fv * c[i] + iv * gv;
      const float hv = ov * tanhf(cv);
      c[i] = cv;
      h[i] = (__bf16)hv;
      const size_t oidx = ((size_t)b * SEQ + t) * HDIM + j;
      if (ys_bf)  ys_bf[oidx]  = (__bf16)hv;
      if (ys_f32) ys_f32[oidx] = hv;
    }
    grid_barrier(bar_cnt, bar_gen, NWG_REC);
  }
}

// ------------------------- small utility kernels ---------------------------
__global__ void k_f32_to_bf16(const float* __restrict__ s, __bf16* __restrict__ d, int n) {
  int i = blockIdx.x * blockDim.x + threadIdx.x;
  if (i < n) d[i] = (__bf16)s[i];
}

__global__ void k_bias_comb(const float* __restrict__ a, const float* __restrict__ b,
                            float* __restrict__ o, int n) {
  int i = blockIdx.x * blockDim.x + threadIdx.x;
  if (i < n) o[i] = a[i] + b[i];
}

__global__ void k_embed_enc(const int* __restrict__ tok, const float* __restrict__ emb,
                            __bf16* __restrict__ X) {
  size_t i = (size_t)blockIdx.x * blockDim.x + threadIdx.x;  // B*S*E elems
  if (i >= (size_t)BATCH * SEQ * EDIM) return;
  int r = (int)(i / EDIM);
  int e = (int)(i % EDIM);
  X[i] = (__bf16)emb[(size_t)tok[r] * EDIM + e];
}

__global__ void k_embed_dec(const int* __restrict__ tgt, const float* __restrict__ emb,
                            __bf16* __restrict__ X) {
  size_t i = (size_t)blockIdx.x * blockDim.x + threadIdx.x;
  if (i >= (size_t)BATCH * SEQ * EDIM) return;
  int r = (int)(i / EDIM);
  int e = (int)(i % EDIM);
  int t = r % SEQ;
  int b = r / SEQ;
  int token = (t == 0) ? BOS_IDX : tgt[b * SEQ + t - 1];
  X[i] = (__bf16)emb[(size_t)token * EDIM + e];
}

// ------------------------- host orchestration ------------------------------
extern "C" void kernel_launch(void* const* d_in, const int* in_sizes, int n_in,
                              void* d_out, int out_size, void* d_ws, size_t ws_size,
                              hipStream_t stream) {
  (void)in_sizes; (void)n_in; (void)out_size; (void)ws_size;

  const int*   src     = (const int*)d_in[0];
  const int*   tgt     = (const int*)d_in[1];
  const float* enc_emb = (const float*)d_in[2];
  const float* dec_emb = (const float*)d_in[3];
  // layer order: enc0, enc1, dec0, dec1
  const float* Wih[4] = {(const float*)d_in[4],  (const float*)d_in[8],
                         (const float*)d_in[12], (const float*)d_in[16]};
  const float* Whh[4] = {(const float*)d_in[5],  (const float*)d_in[9],
                         (const float*)d_in[13], (const float*)d_in[17]};
  const float* bih[4] = {(const float*)d_in[6],  (const float*)d_in[10],
                         (const float*)d_in[14], (const float*)d_in[18]};
  const float* bhh[4] = {(const float*)d_in[7],  (const float*)d_in[11],
                         (const float*)d_in[15], (const float*)d_in[19]};
  const int Kin[4] = {EDIM, HDIM, EDIM, HDIM};

  // ---- workspace carve-up (256B aligned) ----
  char* p = (char*)d_ws;
  auto carve = [&](size_t bytes) -> void* {
    void* r = (void*)p;
    p += (bytes + 255) & ~(size_t)255;
    return r;
  };
  __bf16* Wih_bf[4];
  __bf16* Whh_bf[4];
  float*  bias[4];
  for (int l = 0; l < 4; ++l) {
    Wih_bf[l] = (__bf16*)carve((size_t)GDIM * Kin[l] * sizeof(__bf16));
    Whh_bf[l] = (__bf16*)carve((size_t)GDIM * HDIM * sizeof(__bf16));
    bias[l]   = (float*)carve((size_t)GDIM * sizeof(float));
  }
  __bf16* Xa = (__bf16*)carve((size_t)BATCH * SEQ * HDIM * sizeof(__bf16));
  __bf16* Xb = (__bf16*)carve((size_t)BATCH * SEQ * HDIM * sizeof(__bf16));
  float*  P  = (float*)carve((size_t)BATCH * SEQ * GDIM * sizeof(float));
  float*  Gp = (float*)carve((size_t)NSPLIT * BATCH * GDIM * sizeof(float));
  __bf16* hbf[2];
  float*  cst[2];
  for (int l = 0; l < 2; ++l) {
    hbf[l] = (__bf16*)carve((size_t)BATCH * HDIM * sizeof(__bf16));
    cst[l] = (float*)carve((size_t)BATCH * HDIM * sizeof(float));
  }
  unsigned* bar = (unsigned*)carve(256);   // [0]=counter, [1]=generation

  auto blk = [](size_t n) { return (unsigned)((n + 255) / 256); };

  // ---- weight conversion + bias fold (deterministic, every call) ----
  for (int l = 0; l < 4; ++l) {
    size_t nih = (size_t)GDIM * Kin[l];
    size_t nhh = (size_t)GDIM * HDIM;
    k_f32_to_bf16<<<blk(nih), 256, 0, stream>>>(Wih[l], Wih_bf[l], (int)nih);
    k_f32_to_bf16<<<blk(nhh), 256, 0, stream>>>(Whh[l], Whh_bf[l], (int)nhh);
    k_bias_comb<<<blk(GDIM), 256, 0, stream>>>(bih[l], bhh[l], bias[l], GDIM);
  }
  for (int l = 0; l < 2; ++l) {
    hipMemsetAsync(hbf[l], 0, (size_t)BATCH * HDIM * sizeof(__bf16), stream);
    hipMemsetAsync(cst[l], 0, (size_t)BATCH * HDIM * sizeof(float), stream);
  }
  hipMemsetAsync(bar, 0, 256, stream);

  const int M_big = BATCH * SEQ;                                        // 4096
  const unsigned big_blocks = (unsigned)((M_big / 32) * (GDIM / 32) / 8);  // 2048

  auto run_layer = [&](int wl, int state, const __bf16* Xin, int K,
                       __bf16* ys_bf, float* ys_f32) {
    // P = Xin @ Wih^T + (bih+bhh)   [4096 x 4096]
    k_wmma_gemm<<<big_blocks, 256, 0, stream>>>(
        Xin, K, M_big, Wih_bf[wl], K, bias[wl], /*add_rs=*/0L, P, (long)GDIM, GDIM);
    // whole recurrence in one persistent kernel
    k_lstm_layer<<<NWG_REC, 256, 0, stream>>>(
        Whh_bf[wl], P, Gp, cst[state], hbf[state], ys_bf, ys_f32,
        bar, bar + 1);
  };

  // ---- encoder ----
  k_embed_enc<<<blk((size_t)BATCH * SEQ * EDIM), 256, 0, stream>>>(src, enc_emb, Xa);
  run_layer(/*wl=*/0, /*state=*/0, Xa, EDIM, /*ys_bf=*/Xb, /*ys_f32=*/nullptr);
  run_layer(/*wl=*/1, /*state=*/1, Xb, HDIM, /*ys_bf=*/nullptr, /*ys_f32=*/nullptr);

  // ---- decoder (h0/c0 = encoder final states, already in hbf/cst) ----
  k_embed_dec<<<blk((size_t)BATCH * SEQ * EDIM), 256, 0, stream>>>(tgt, dec_emb, Xa);
  run_layer(/*wl=*/2, /*state=*/0, Xa, EDIM, /*ys_bf=*/Xb, /*ys_f32=*/nullptr);
  run_layer(/*wl=*/3, /*state=*/1, Xb, HDIM, /*ys_bf=*/nullptr, /*ys_f32=*/(float*)d_out);
}
